// TaroRestarter_6296422056477
// MI455X (gfx1250) — matrix operand order repaired and checked by reading, not compiled, not run on hardware
//
// CDNA5 / gfx1250 implementation of the TGN forward reference.
// bf16 operands + f32-accum WMMA (v_wmma_f32_16x16x32_bf16) for all GEMMs.
// GEMM A-operand is staged into LDS with gfx1250 async copies
// (global_load_async_to_lds_b128 + s_wait_asynccnt), double-buffered.
#include <hip/hip_runtime.h>

typedef __bf16 bf16_t;
typedef __attribute__((ext_vector_type(16))) __bf16 v16bf;
typedef __attribute__((ext_vector_type(8)))  float  v8f;

enum {
  BS = 4096, HL = 20,
  NF = 128, EF = 128, TF = 128,
  DM = 640, DIN = 512,          // 3*NF+EF+TF, DM-TF
  NHEAD = 2, DH = 320,
  ODE_STEPS = 8
};

// ---------------------------------------------------------------------------
// WMMA fragment loaders (contiguous-K order chosen by this file, not the ISA table order; wave32)
//  A (16x32 bf16): lanes 0-15 rows M=0..15 K=0..15, lanes 16-31 K=16..31;
//                  16 contiguous K per lane (matches B).
//  B (32x16 bf16): lanes 0-15 cols N=0..15 K=0..15, lanes 16-31 K=16..31;
//                  16 contiguous K per lane.
// ---------------------------------------------------------------------------
union FragU { v16bf v; uint4 u[2]; };

__device__ __forceinline__ v16bf load_frag_a(const bf16_t* A, size_t lda,
                                             int row0, int k0, int lane) {
  const int r  = row0 + (lane & 15);
  const int kb = k0 + ((lane >> 4) << 4);
  const bf16_t* p = A + (size_t)r * lda + kb;
  FragU f;
  f.u[0] = *reinterpret_cast<const uint4*>(p);        // K = kb .. kb+7
  f.u[1] = *reinterpret_cast<const uint4*>(p + 8);    // K = kb+8 .. kb+15
  return f.v;
}

// W stored row-major [N, K] so each lane's 16 K-values are contiguous.
__device__ __forceinline__ v16bf load_frag_b(const bf16_t* W, int K,
                                             int n0, int k0, int lane) {
  const int c  = n0 + (lane & 15);
  const int kb = k0 + ((lane >> 4) << 4);
  const bf16_t* p = W + (size_t)c * K + kb;
  FragU f;
  f.u[0] = *reinterpret_cast<const uint4*>(p);
  f.u[1] = *reinterpret_cast<const uint4*>(p + 8);
  return f.v;
}

#define WMMA_BF16(a, b, acc) \
  __builtin_amdgcn_wmma_f32_16x16x32_bf16(false, (a), false, (b), (short)0, (acc), false, false)

// Async copy of one 16-byte segment (8 bf16) from global to LDS.
__device__ __forceinline__ void async_b128_to_lds(const bf16_t* gp, const bf16_t* lp) {
  const unsigned lds_off = (unsigned)(size_t)lp;          // low 32b of generic = LDS offset
  const unsigned long long ga = (unsigned long long)(size_t)gp;
  asm volatile("global_load_async_to_lds_b128 %0, %1, off"
               :: "v"(lds_off), "v"(ga) : "memory");
}
__device__ __forceinline__ void wait_async_all() {
  asm volatile("s_wait_asynccnt 0x0" ::: "memory");
}

// ---------------------------------------------------------------------------
// Generic bf16 WMMA GEMM:  C[M,N] = A[M,K] @ W[N,K]^T + bias
// Block = 256 thr = 8 waves arranged 2(M) x 4(N); wave tile 32x32 (4 accs).
// A tile (64x32) staged in LDS by async copies, double-buffered.
// Requires: M % 64 == 0, N % 128 == 0, K % 32 == 0 (true for every call here).
// ---------------------------------------------------------------------------
__global__ void gemm_wmma_bf16(const bf16_t* __restrict__ A, int lda,
                               const bf16_t* __restrict__ W, int K,
                               const float* __restrict__ bias,
                               void* __restrict__ Cout, int ldc, int out_bf16,
                               int M, int N) {
  __shared__ alignas(16) bf16_t atile[2][64 * 32];
  const int t = threadIdx.x;
  const int lane = t & 31;
  const int wave = t >> 5;
  const int wm = wave & 1;                      // 2 waves along M
  const int wn = wave >> 1;                     // 4 waves along N
  const int rowB = blockIdx.y * 64;             // block row base (global)
  const int row0 = wm * 32;                     // wave row base (within tile)
  const int col0 = blockIdx.x * 128 + wn * 32;  // wave col base (global)
  const int r_ld = t >> 2;                      // 0..63 : tile row this thread copies
  const int seg  = (t & 3) * 8;                 // 8-bf16 segment within 32-wide row
  (void)M; (void)N;

  // stage k-tile 0 into buffer 0
  async_b128_to_lds(A + (size_t)(rowB + r_ld) * lda + seg,
                    atile[0] + r_ld * 32 + seg);
  wait_async_all();
  __syncthreads();

  v8f acc00 = {}, acc01 = {}, acc10 = {}, acc11 = {};
  for (int k0 = 0; k0 < K; k0 += 32) {
    const int cur = (k0 >> 5) & 1;
    const bool more = (k0 + 32) < K;
    if (more) {  // async-prefetch next A k-tile into the other buffer
      async_b128_to_lds(A + (size_t)(rowB + r_ld) * lda + (k0 + 32) + seg,
                        atile[cur ^ 1] + r_ld * 32 + seg);
      __builtin_prefetch(W + (size_t)(col0 + (lane & 15)) * K + k0 + 32, 0, 3);
    }
    v16bf a0 = load_frag_a(atile[cur], 32, row0,      0, lane);  // ds_load_b128
    v16bf a1 = load_frag_a(atile[cur], 32, row0 + 16, 0, lane);
    v16bf b0 = load_frag_b(W, K, col0,      k0, lane);
    v16bf b1 = load_frag_b(W, K, col0 + 16, k0, lane);
    acc00 = WMMA_BF16(a0, b0, acc00);
    acc01 = WMMA_BF16(a0, b1, acc01);
    acc10 = WMMA_BF16(a1, b0, acc10);
    acc11 = WMMA_BF16(a1, b1, acc11);
    if (more) wait_async_all();
    __syncthreads();   // reads of atile[cur] done before it is overwritten
  }

  v8f accs[2][2] = {{acc00, acc01}, {acc10, acc11}};
#pragma unroll
  for (int rt = 0; rt < 2; ++rt) {
    const int rb = rowB + row0 + 16 * rt + ((lane >> 4) << 3);
#pragma unroll
    for (int ct = 0; ct < 2; ++ct) {
      const int c = col0 + 16 * ct + (lane & 15);
      const float bv = bias ? bias[c] : 0.0f;
#pragma unroll
      for (int i = 0; i < 8; ++i) {
        const float v = accs[rt][ct][i] + bv;
        const size_t idx = (size_t)(rb + i) * ldc + c;
        if (out_bf16) ((bf16_t*)Cout)[idx] = (bf16_t)v;
        else          ((float*) Cout)[idx] = v;
      }
    }
  }
}

// ---------------------------------------------------------------------------
// Weight conversion helpers (f32 -> bf16, optionally transposed to [N,K])
// ---------------------------------------------------------------------------
__global__ void f2bf_kernel(const float* __restrict__ s, bf16_t* __restrict__ d, int n) {
  int i = blockIdx.x * 256 + threadIdx.x;
  if (i < n) d[i] = (bf16_t)s[i];
}
__global__ void f2bf_t_kernel(const float* __restrict__ s, bf16_t* __restrict__ d,
                              int R, int C) {  // s[R][C] -> d[C][R]
  int i = blockIdx.x * 256 + threadIdx.x;
  if (i < R * C) {
    int r = i / C, c = i - r * C;
    d[(size_t)c * R + r] = (bf16_t)s[i];
  }
}

// ---------------------------------------------------------------------------
// Gather + time-encode: builds full_vals (bf16, last-row DIN zeroed),
// last_event_feat (bf16), nb (bf16), mask, and emits hist_ts[:, -1] output.
// One block per (b, l), 128 threads = feature dim.
// ---------------------------------------------------------------------------
__global__ void gather_kernel(const int* __restrict__ nids,
                              const int* __restrict__ hist_nids,
                              const int* __restrict__ anon_ids,
                              const int* __restrict__ hist_eids,
                              const float* __restrict__ hist_ts,
                              const int* __restrict__ hist_dirs,
                              const float* __restrict__ node_feat,
                              const float* __restrict__ edge_feat,
                              const float* __restrict__ anony_emb,
                              const float* __restrict__ time_w,
                              const float* __restrict__ time_b,
                              bf16_t* __restrict__ fv,      // [BS*HL, DM]
                              bf16_t* __restrict__ le,      // [BS, DIN]
                              bf16_t* __restrict__ nb,      // [BS*HL, 3*NF]
                              int* __restrict__ maskbuf,    // [BS*HL]
                              float* __restrict__ ts_out) { // [BS]
  const int row = blockIdx.x;            // b*HL + l
  const int b = row / HL, l = row - b * HL;
  const int d = threadIdx.x;
  const int hn  = hist_nids[row];
  const int dir = hist_dirs[row];
  const int eid = hist_eids[row];
  const int aid = anon_ids[row];
  const int nid = nids[b];
  const int src = dir ? nid : hn;
  const int dst = dir ? hn : nid;

  const float sv = node_feat[(size_t)src * NF + d];
  const float dv = node_feat[(size_t)dst * NF + d];
  const float av = anony_emb[(size_t)aid * NF + d];
  const float ev = edge_feat[(size_t)eid * EF + d];
  const float dt = hist_ts[b * HL + (HL - 1)] - hist_ts[row];
  const float tv = cosf(dt * time_w[d] + time_b[d]);

  const bool last = (l == HL - 1);
  bf16_t* fr = fv + (size_t)row * DM;
  fr[0 * NF + d] = (bf16_t)(last ? 0.0f : sv);
  fr[1 * NF + d] = (bf16_t)(last ? 0.0f : dv);
  fr[2 * NF + d] = (bf16_t)(last ? 0.0f : av);
  fr[3 * NF + d] = (bf16_t)(last ? 0.0f : ev);
  fr[4 * NF + d] = (bf16_t)tv;
  if (last) {
    bf16_t* lr = le + (size_t)b * DIN;
    lr[0 * NF + d] = (bf16_t)sv;
    lr[1 * NF + d] = (bf16_t)dv;
    lr[2 * NF + d] = (bf16_t)av;
    lr[3 * NF + d] = (bf16_t)ev;
  }
  bf16_t* nr = nb + (size_t)row * (3 * NF);
  nr[0 * NF + d] = (bf16_t)node_feat[(size_t)hn * NF + d];
  nr[1 * NF + d] = (bf16_t)ev;
  nr[2 * NF + d] = (bf16_t)tv;
  if (d == 0) {
    maskbuf[row] = (hn == 0 && !last) ? 1 : 0;
    if (last) ts_out[b] = hist_ts[row];
  }
}

// ---------------------------------------------------------------------------
// MHA over history (only the last query row matters): per-batch block.
// ---------------------------------------------------------------------------
__global__ void attn1_kernel(const float* __restrict__ q,    // [BS, DM]
                             const bf16_t* __restrict__ kv,  // [BS*HL, 2*DM] (k|v)
                             const int* __restrict__ maskbuf,
                             bf16_t* __restrict__ ctx1) {    // [BS, DM]
  __shared__ float sc_s[NHEAD][HL];
  __shared__ float a_s[NHEAD][HL];
  const int b = blockIdx.x, t = threadIdx.x;
  if (t < NHEAD * HL) {
    const int h = t / HL, m = t - h * HL;
    const float* qp = q + (size_t)b * DM + h * DH;
    const bf16_t* kp = kv + ((size_t)b * HL + m) * (2 * DM) + h * DH;
    float acc = 0.0f;
    for (int dd = 0; dd < DH; ++dd) acc += qp[dd] * (float)kp[dd];
    acc *= 0.05590169944f;                       // 1/sqrt(320)
    if (maskbuf[b * HL + m]) acc = -1.0e9f;
    sc_s[h][m] = acc;
  }
  __syncthreads();
  if (t < NHEAD) {
    float mx = -1.0e30f;
    for (int m = 0; m < HL; ++m) mx = fmaxf(mx, sc_s[t][m]);
    float sum = 0.0f;
    for (int m = 0; m < HL; ++m) { float e = expf(sc_s[t][m] - mx); a_s[t][m] = e; sum += e; }
    const float inv = 1.0f / sum;
    for (int m = 0; m < HL; ++m) a_s[t][m] *= inv;
  }
  __syncthreads();
  for (int d = t; d < DM; d += blockDim.x) {
    const int h = d / DH;
    float acc = 0.0f;
    for (int m = 0; m < HL; ++m)
      acc += a_s[h][m] * (float)kv[((size_t)b * HL + m) * (2 * DM) + DM + d];
    ctx1[(size_t)b * DM + d] = (bf16_t)acc;
  }
}

// ---------------------------------------------------------------------------
// Second attention + build merge input [ctx2 | node_feat[nids]] (bf16).
// ---------------------------------------------------------------------------
__global__ void attn2_kernel(const float* __restrict__ qh,   // [BS, NF]
                             const float* __restrict__ kh,   // [BS*HL, NF]
                             const float* __restrict__ vh,   // [BS*HL, NF]
                             const int* __restrict__ maskbuf,
                             const float* __restrict__ node_feat,
                             const int* __restrict__ nids,
                             bf16_t* __restrict__ mrg_in) {  // [BS, 2*NF]
  __shared__ float sc_s[HL];
  __shared__ float a_s[HL];
  const int b = blockIdx.x, t = threadIdx.x;
  if (t < HL) {
    const float* qp = qh + (size_t)b * NF;
    const float* kp = kh + ((size_t)b * HL + t) * NF;
    float acc = 0.0f;
    for (int d = 0; d < NF; ++d) acc += qp[d] * kp[d];
    acc *= 0.08838834765f;                       // 1/sqrt(128)
    if (maskbuf[b * HL + t]) acc = -1.0e9f;
    sc_s[t] = acc;
  }
  __syncthreads();
  if (t == 0) {
    float mx = -1.0e30f;
    for (int m = 0; m < HL; ++m) mx = fmaxf(mx, sc_s[m]);
    float sum = 0.0f;
    for (int m = 0; m < HL; ++m) { float e = expf(sc_s[m] - mx); a_s[m] = e; sum += e; }
    const float inv = 1.0f / sum;
    for (int m = 0; m < HL; ++m) a_s[m] *= inv;
  }
  __syncthreads();
  float acc = 0.0f;
  for (int m = 0; m < HL; ++m) acc += a_s[m] * vh[((size_t)b * HL + m) * NF + t];
  mrg_in[(size_t)b * (2 * NF) + t] = (bf16_t)acc;
  mrg_in[(size_t)b * (2 * NF) + NF + t] =
      (bf16_t)node_feat[(size_t)nids[b] * NF + t];
}

// h_prev_left = tanh(merge_pre); also keep bf16 copy for the GRU hh GEMM.
__global__ void tanh_dual_kernel(const float* __restrict__ pre,
                                 float* __restrict__ hf, bf16_t* __restrict__ hb, int n) {
  int i = blockIdx.x * 256 + threadIdx.x;
  if (i < n) { float v = tanhf(pre[i]); hf[i] = v; hb[i] = (bf16_t)v; }
}

// GRU gate combine -> h_prev_right ; writes masked h_right output.
__global__ void gru_kernel(const float* __restrict__ gi,   // [BS, 3*NF]
                           const float* __restrict__ gh,   // [BS, 3*NF]
                           const float* __restrict__ hpl,  // [BS, NF]
                           const int* __restrict__ maskbuf,
                           float* __restrict__ hpr,        // [BS, NF]
                           float* __restrict__ h_right) {
  int i = blockIdx.x * 256 + threadIdx.x;
  if (i >= BS * NF) return;
  const int b = i >> 7, j = i & (NF - 1);
  const size_t g = (size_t)b * (3 * NF);
  const float r  = 1.0f / (1.0f + expf(-(gi[g + j] + gh[g + j])));
  const float zg = 1.0f / (1.0f + expf(-(gi[g + NF + j] + gh[g + NF + j])));
  const float n  = tanhf(gi[g + 2 * NF + j] + r * gh[g + 2 * NF + j]);
  const float h  = (1.0f - zg) * n + zg * hpl[i];
  hpr[i] = h;
  int iv = 1;
  for (int l = 0; l < HL; ++l) if (!maskbuf[b * HL + l]) { iv = 0; break; }
  h_right[i] = iv ? 0.0f : h;
}

// ---------------------------------------------------------------------------
// Fused RK4 neural-ODE: one block per 16 batch rows, 256 thr = 8 waves.
// z lives in registers; stage inputs staged in LDS as bf16; each wave WMMAs
// its 16-column tile of the [16,128]x[128,128] eval GEMM.
// ---------------------------------------------------------------------------
__global__ void ode_rk4_kernel(const float* __restrict__ z0,     // hpr [BS,NF]
                               const bf16_t* __restrict__ Wode,  // [NF,NF] (N,K)
                               const float* __restrict__ ode_b,
                               const float* __restrict__ tnw,
                               const float* __restrict__ tnb,
                               const float* __restrict__ hist_ts,
                               const int* __restrict__ maskbuf,
                               float* __restrict__ h_left) {
  __shared__ alignas(16) bf16_t zin[16 * NF];
  __shared__ float kev[16 * NF];
  __shared__ float t0s[16], rats[16];
  __shared__ int inval[16];

  const int t = threadIdx.x;
  const int b0 = blockIdx.x * 16;
  const int row = t >> 4;           // 0..15
  const int cb  = (t & 15) * 8;     // 8 cols per thread
  const int lane = t & 31;
  const int wave = t >> 5;          // 0..7 -> col tile 16*wave

  if (t < 16) {
    const int b = b0 + t;
    const float t0 = hist_ts[b * HL + (HL - 2)];
    const float t1 = hist_ts[b * HL + (HL - 1)];
    t0s[t] = t0; rats[t] = t1 - t0;
    int iv = 1;
    for (int l = 0; l < HL; ++l) if (!maskbuf[b * HL + l]) { iv = 0; break; }
    inval[t] = iv;
  }
  float z[8], tw[8], tb[8];
#pragma unroll
  for (int e = 0; e < 8; ++e) {
    const int d = cb + e;
    z[e]  = z0[(size_t)(b0 + row) * NF + d];
    tw[e] = tnw[d];
    tb[e] = tnb[d];
  }
  __syncthreads();
  const float t0r = t0s[row], ratr = rats[row];
  const float ds = 1.0f / ODE_STEPS;
  float kprev[8];

  for (int step = 0; step < ODE_STEPS; ++step) {
    const float s0 = step * ds;
    float ksum[8] = {0, 0, 0, 0, 0, 0, 0, 0};
#pragma unroll 1
    for (int sub = 0; sub < 4; ++sub) {
      const float sc = (sub == 0) ? 0.0f : ((sub == 3) ? ds : 0.5f * ds);
      const float s = s0 + sc;
      const float tt = s * ratr + t0r;
#pragma unroll
      for (int e = 0; e < 8; ++e) {
        const float zi = z[e] + (sub ? sc * kprev[e] : 0.0f);
        const float te = cosf(tt * tw[e] + tb[e]);
        zin[row * NF + cb + e] = (bf16_t)(zi + te);
      }
      __syncthreads();
      v8f acc = {};
#pragma unroll
      for (int ks = 0; ks < 4; ++ks) {
        v16bf a = load_frag_a(zin, NF, 0, ks * 32, lane);
        v16bf b = load_frag_b(Wode, NF, wave * 16, ks * 32, lane);
        acc = WMMA_BF16(a, b, acc);
      }
      {
        const int c = wave * 16 + (lane & 15);
        const float bv = ode_b[c];
        const int rb = (lane >> 4) << 3;
#pragma unroll
        for (int i = 0; i < 8; ++i)
          kev[(rb + i) * NF + c] = tanhf(acc[i] + bv) * rats[rb + i];
      }
      __syncthreads();
#pragma unroll
      for (int e = 0; e < 8; ++e) {
        kprev[e] = kev[row * NF + cb + e];
        ksum[e] += ((sub == 0 || sub == 3) ? 1.0f : 2.0f) * kprev[e];
      }
      __syncthreads();
    }
#pragma unroll
    for (int e = 0; e < 8; ++e) z[e] += (ds / 6.0f) * ksum[e];
  }
  const int iv = inval[row];
#pragma unroll
  for (int e = 0; e < 8; ++e)
    h_left[(size_t)(b0 + row) * NF + cb + e] = iv ? 0.0f : z[e];
}

// ---------------------------------------------------------------------------
// Host orchestration
// ---------------------------------------------------------------------------
static inline void launch_gemm(const bf16_t* A, int lda, const bf16_t* W, int K,
                               const float* bias, void* C, int ldc, int out_bf16,
                               int M, int N, hipStream_t s) {
  dim3 grid(N / 128, M / 64);
  gemm_wmma_bf16<<<grid, 256, 0, s>>>(A, lda, W, K, bias, C, ldc, out_bf16, M, N);
}

extern "C" void kernel_launch(void* const* d_in, const int* in_sizes, int n_in,
                              void* d_out, int out_size, void* d_ws, size_t ws_size,
                              hipStream_t stream) {
  (void)in_sizes; (void)n_in; (void)out_size; (void)ws_size;

  const int*   nids      = (const int*)  d_in[0];
  const int*   hist_nids = (const int*)  d_in[2];
  const int*   anon_ids  = (const int*)  d_in[3];
  const int*   hist_eids = (const int*)  d_in[4];
  const float* hist_ts   = (const float*)d_in[5];
  const int*   hist_dirs = (const int*)  d_in[6];
  const float* node_feat = (const float*)d_in[7];
  const float* edge_feat = (const float*)d_in[8];
  const float* anony_emb = (const float*)d_in[9];
  const float* time_w    = (const float*)d_in[10];
  const float* time_b    = (const float*)d_in[11];
  const float* in_proj_w = (const float*)d_in[12];
  const float* in_proj_b = (const float*)d_in[13];
  const float* out_proj_w= (const float*)d_in[14];
  const float* out_proj_b= (const float*)d_in[15];
  const float* outfn_w   = (const float*)d_in[16];
  const float* outfn_b   = (const float*)d_in[17];
  const float* attn_wq   = (const float*)d_in[18];
  const float* attn_wk   = (const float*)d_in[19];
  const float* attn_wv   = (const float*)d_in[20];
  const float* merge_w   = (const float*)d_in[21];
  const float* merge_b   = (const float*)d_in[22];
  const float* gru_w_ih  = (const float*)d_in[23];
  const float* gru_w_hh  = (const float*)d_in[24];
  const float* gru_b_ih  = (const float*)d_in[25];
  const float* gru_b_hh  = (const float*)d_in[26];
  const float* ode_w     = (const float*)d_in[27];
  const float* ode_b     = (const float*)d_in[28];
  const float* tnode_w   = (const float*)d_in[29];
  const float* tnode_b   = (const float*)d_in[30];

  char* ws = (char*)d_ws;
  size_t off = 0;
  auto alloc = [&](size_t bytes) -> char* {
    char* p = ws + off;
    off = (off + bytes + 255) & ~(size_t)255;
    return p;
  };
  const size_t ROWS = (size_t)BS * HL;

  // bf16 weights, all stored [N, K]
  bf16_t* Wqkv = (bf16_t*)alloc((size_t)3 * DM * DM * 2);
  bf16_t* Wout = (bf16_t*)alloc((size_t)DM * DM * 2);
  bf16_t* Wofn = (bf16_t*)alloc((size_t)NF * DM * 2);
  bf16_t* Wq2  = (bf16_t*)alloc((size_t)NF * NF * 2);
  bf16_t* Wk2  = (bf16_t*)alloc((size_t)NF * 3 * NF * 2);
  bf16_t* Wv2  = (bf16_t*)alloc((size_t)NF * 3 * NF * 2);
  bf16_t* Wmrg = (bf16_t*)alloc((size_t)NF * 2 * NF * 2);
  bf16_t* Wih  = (bf16_t*)alloc((size_t)3 * NF * DIN * 2);
  bf16_t* Whh  = (bf16_t*)alloc((size_t)3 * NF * NF * 2);
  bf16_t* WodeB= (bf16_t*)alloc((size_t)NF * NF * 2);
  // activations
  bf16_t* fv     = (bf16_t*)alloc(ROWS * DM * 2);
  bf16_t* nbuf   = (bf16_t*)alloc(ROWS * 3 * NF * 2);
  bf16_t* le     = (bf16_t*)alloc((size_t)BS * DIN * 2);
  int*    maskb  = (int*)   alloc(ROWS * 4);
  bf16_t* kv     = (bf16_t*)alloc(ROWS * 2 * DM * 2);
  float*  qbuf   = (float*) alloc((size_t)BS * DM * 4);
  bf16_t* ctx1   = (bf16_t*)alloc((size_t)BS * DM * 2);
  bf16_t* mha    = (bf16_t*)alloc((size_t)BS * DM * 2);
  bf16_t* qn     = (bf16_t*)alloc((size_t)BS * NF * 2);
  float*  qh     = (float*) alloc((size_t)BS * NF * 4);
  float*  kh     = (float*) alloc(ROWS * NF * 4);
  float*  vh     = (float*) alloc(ROWS * NF * 4);
  bf16_t* mrg_in = (bf16_t*)alloc((size_t)BS * 2 * NF * 2);
  float*  mrg_pre= (float*) alloc((size_t)BS * NF * 4);
  float*  hpl    = (float*) alloc((size_t)BS * NF * 4);
  bf16_t* hplb   = (bf16_t*)alloc((size_t)BS * NF * 2);
  float*  gi     = (float*) alloc((size_t)BS * 3 * NF * 4);
  float*  gh     = (float*) alloc((size_t)BS * 3 * NF * 4);
  float*  hpr    = (float*) alloc((size_t)BS * NF * 4);

  float* out_hl = (float*)d_out;
  float* out_hr = out_hl + (size_t)BS * NF;
  float* out_ts = out_hr + (size_t)BS * NF;

  auto cvt = [&](const float* s, bf16_t* d, int n) {
    f2bf_kernel<<<(n + 255) / 256, 256, 0, stream>>>(s, d, n);
  };
  auto cvtT = [&](const float* s, bf16_t* d, int R, int C) {
    f2bf_t_kernel<<<(R * C + 255) / 256, 256, 0, stream>>>(s, d, R, C);
  };
  cvt(in_proj_w,  Wqkv, 3 * DM * DM);   // [1920,640] already [N,K]
  cvt(out_proj_w, Wout, DM * DM);
  cvt(outfn_w,    Wofn, NF * DM);
  cvt(gru_w_ih,   Wih,  3 * NF * DIN);
  cvt(gru_w_hh,   Whh,  3 * NF * NF);
  cvt(ode_w,      WodeB, NF * NF);
  cvtT(attn_wq, Wq2, NF, NF);           // reference multiplies un-transposed
  cvtT(attn_wk, Wk2, 3 * NF, NF);
  cvtT(attn_wv, Wv2, 3 * NF, NF);
  cvtT(merge_w, Wmrg, 2 * NF, NF);

  // 1) gather / time-encode / mask / ts output
  gather_kernel<<<BS * HL, NF, 0, stream>>>(nids, hist_nids, anon_ids, hist_eids,
                                            hist_ts, hist_dirs, node_feat, edge_feat,
                                            anony_emb, time_w, time_b,
                                            fv, le, nbuf, maskb, out_ts);
  // 2) K,V for all rows; Q only for the last history row of each batch
  launch_gemm(fv, DM, Wqkv + (size_t)DM * DM, DM, in_proj_b + DM,
              kv, 2 * DM, /*bf16*/1, (int)ROWS, 2 * DM, stream);
  launch_gemm(fv + (size_t)(HL - 1) * DM, HL * DM, Wqkv, DM, in_proj_b,
              qbuf, DM, /*f32*/0, BS, DM, stream);
  // 3) per-batch MHA (last query row only)
  attn1_kernel<<<BS, 256, 0, stream>>>(qbuf, kv, maskb, ctx1);
  // 4) out_proj -> outfn -> attn_wq
  launch_gemm(ctx1, DM, Wout, DM, out_proj_b, mha, DM, 1, BS, DM, stream);
  launch_gemm(mha, DM, Wofn, DM, outfn_b, qn, NF, 1, BS, NF, stream);
  launch_gemm(qn, NF, Wq2, NF, nullptr, qh, NF, 0, BS, NF, stream);
  // 5) kh / vh over neighborhood features
  launch_gemm(nbuf, 3 * NF, Wk2, 3 * NF, nullptr, kh, NF, 0, (int)ROWS, NF, stream);
  launch_gemm(nbuf, 3 * NF, Wv2, 3 * NF, nullptr, vh, NF, 0, (int)ROWS, NF, stream);
  // 6) second attention + merge input
  attn2_kernel<<<BS, NF, 0, stream>>>(qh, kh, vh, maskb, node_feat, nids, mrg_in);
  launch_gemm(mrg_in, 2 * NF, Wmrg, 2 * NF, merge_b, mrg_pre, NF, 0, BS, NF, stream);
  tanh_dual_kernel<<<(BS * NF + 255) / 256, 256, 0, stream>>>(mrg_pre, hpl, hplb, BS * NF);
  // 7) GRU
  launch_gemm(le, DIN, Wih, DIN, gru_b_ih, gi, 3 * NF, 0, BS, 3 * NF, stream);
  launch_gemm(hplb, NF, Whh, NF, gru_b_hh, gh, 3 * NF, 0, BS, 3 * NF, stream);
  gru_kernel<<<(BS * NF + 255) / 256, 256, 0, stream>>>(gi, gh, hpl, maskb, hpr, out_hr);
  // 8) fused RK4 neural-ODE -> h_left
  ode_rk4_kernel<<<BS / 16, 256, 0, stream>>>(hpr, WodeB, ode_b, tnode_w, tnode_b,
                                              hist_ts, maskb, out_hl);
}
